// KGVAE_90426241450704
// MI455X (gfx1250) — matrix operand order repaired and checked by reading, not compiled, not run on hardware
//
#include <hip/hip_runtime.h>

#define N_NODES 20000
#define N_PAD   20096   // N_NODES padded up to a multiple of 128 for decoder B
#define N_EDGES 256000
#define HD      128
#define HD2     256
#define NB      64
#define B_Q     2048

typedef __bf16 bf16_t;
typedef __attribute__((ext_vector_type(16))) __bf16 v16bf;
typedef __attribute__((ext_vector_type(8)))  __bf16 v8bf;
typedef __attribute__((ext_vector_type(8)))  float  v8f;

// ---------------------------------------------------------------------------
// WMMA bf16 fragment loaders (CDNA5 wave32 layouts, CDNA5 ISA §7.12.2)
// All loads are contiguous 16B vectors, no predication in the hot loop;
// conversion to bf16 happens in producer kernels.
// ---------------------------------------------------------------------------

// A 16x32 bf16 from row-major bf16 [.,ld]: lane(0-15)=rows, K {kh*8..+7, 16+kh*8..+7}
__device__ __forceinline__ v16bf load_a_frag(const bf16_t* __restrict__ A, int ld,
                                             int row_base, int k_base) {
    const int lane = threadIdx.x & 31;
    const int row  = row_base + (lane & 15);
    const int kh   = lane >> 4;
    const bf16_t* __restrict__ p = A + (long)row * ld + k_base + kh * 8;
    v8bf lo = *(const v8bf*)p;          // K = k_base + kh*8 .. +7
    v8bf hi = *(const v8bf*)(p + 16);   // K = k_base + 16 + kh*8 .. +7
    return __builtin_shufflevector(lo, hi, 0,1,2,3,4,5,6,7,8,9,10,11,12,13,14,15);
}

// B 32x16 bf16 from K-contiguous storage BT[col][k] (i.e. B^T row-major, ld=ldk):
// lane col fixed, K {kh*8..+7, 16+kh*8..+7} -> matches A fragment order.
__device__ __forceinline__ v16bf load_b_frag(const bf16_t* __restrict__ BT, int ldk,
                                             int col_base, int k_base) {
    const int lane = threadIdx.x & 31;
    const int col  = col_base + (lane & 15);
    const int kh   = lane >> 4;
    const bf16_t* __restrict__ p = BT + (long)col * ldk + k_base + kh * 8;
    v8bf lo = *(const v8bf*)p;
    v8bf hi = *(const v8bf*)(p + 16);
    return __builtin_shufflevector(lo, hi, 0,1,2,3,4,5,6,7,8,9,10,11,12,13,14,15);
}

// ---------------------------------------------------------------------------
// Elementwise / prep / scatter kernels
// ---------------------------------------------------------------------------
__global__ void k_zero(float* __restrict__ p, long n) {
    long i = (long)blockIdx.x * blockDim.x + threadIdx.x;
    long s = (long)gridDim.x * blockDim.x;
    for (; i < n; i += s) p[i] = 0.0f;
}

// Transpose + convert weight matrix W[K][Nc] (fp32) -> WT[Nc][K] (bf16)
__global__ void k_wT(const float* __restrict__ W, bf16_t* __restrict__ WT,
                     int K, int Nc) {
    int i = blockIdx.x * blockDim.x + threadIdx.x;
    if (i >= K * Nc) return;
    int k = i / Nc, n = i - k * Nc;
    WT[(long)n * K + k] = (bf16_t)W[i];
}

__global__ void k_gather(const float* __restrict__ embed, const int* __restrict__ h,
                         float* __restrict__ x0, bf16_t* __restrict__ x0b) {
    int i = blockIdx.x * blockDim.x + threadIdx.x;
    if (i >= N_NODES * HD) return;
    int n = i >> 7, j = i & (HD - 1);
    float v = embed[(long)h[n] * HD + j];
    x0[i]  = v;
    x0b[i] = (bf16_t)v;
}

// Layer-1 messages: msg[2b+o] = (x[s][2b]*W[b][0][o] + x[s][2b+1]*W[b][1][o]) * norm
__global__ void __launch_bounds__(256) k_edge1(const float* __restrict__ x0,
                                               const int* __restrict__ src,
                                               const int* __restrict__ dst,
                                               const int* __restrict__ et,
                                               const float* __restrict__ norm,
                                               const float* __restrict__ W1,
                                               float* __restrict__ agg1) {
    int t = threadIdx.x;
    int e = blockIdx.x * 2 + (t >> 7);
    if (e >= N_EDGES) return;
    int j = t & (HD - 1);
    int s = src[e], d = dst[e], r = et[e];
    float nm = norm[e];
    int b = j >> 1, o = j & 1;
    const float* w = W1 + ((long)r * NB + b) * 4;       // [2][2]
    float xi0 = x0[(long)s * HD + 2 * b];
    float xi1 = x0[(long)s * HD + 2 * b + 1];
    atomicAdd(&agg1[(long)d * HD + j], (xi0 * w[o] + xi1 * w[2 + o]) * nm);
}

// Layer-2 messages: 256 outputs/edge, blocks of [2][4]
__global__ void __launch_bounds__(256) k_edge2(const float* __restrict__ h1,
                                               const int* __restrict__ src,
                                               const int* __restrict__ dst,
                                               const int* __restrict__ et,
                                               const float* __restrict__ norm,
                                               const float* __restrict__ W2,
                                               float* __restrict__ agg2) {
    int e = blockIdx.x;
    int j = threadIdx.x;                                 // 0..255
    int s = src[e], d = dst[e], r = et[e];
    float nm = norm[e];
    int b = j >> 2, o = j & 3;
    const float* w = W2 + ((long)r * NB + b) * 8;       // [2][4]
    float xi0 = h1[(long)s * HD + 2 * b];
    float xi1 = h1[(long)s * HD + 2 * b + 1];
    atomicAdd(&agg2[(long)d * HD2 + j], (xi0 * w[o] + xi1 * w[4 + o]) * nm);
}

// ---------------------------------------------------------------------------
// WMMA kernels (operands pre-converted bf16, fp32 accumulate)
// ---------------------------------------------------------------------------
// h1 = relu(agg1 + x0 @ lw1 + b1)   [N,128]; 8 waves/block, one 16x16 tile each
__global__ void __launch_bounds__(256) k_l1(const bf16_t* __restrict__ x0b,
                                            const bf16_t* __restrict__ lw1T,
                                            const float* __restrict__ b1,
                                            const float* __restrict__ agg1,
                                            float* __restrict__ h1,
                                            bf16_t* __restrict__ h1b) {
    const int w     = threadIdx.x >> 5;
    const int lane  = threadIdx.x & 31;
    const int mbase = blockIdx.x * 16;
    const int nbase = w * 16;
    v8f acc = {};
#pragma unroll
    for (int kc = 0; kc < HD; kc += 32) {
        v16bf a = load_a_frag(x0b, HD, mbase, kc);
        v16bf b = load_b_frag(lw1T, HD, nbase, kc);
        acc = __builtin_amdgcn_wmma_f32_16x16x32_bf16(false, a, false, b,
                                                      (short)0, acc, false, false);
    }
    const int col   = nbase + (lane & 15);
    const int rbase = mbase + ((lane >> 4) << 3);
    float bias = b1[col];
#pragma unroll
    for (int r = 0; r < 8; ++r) {
        long idx = (long)(rbase + r) * HD + col;
        float v = acc[r] + agg1[idx] + bias;
        v = v > 0.0f ? v : 0.0f;
        h1[idx]  = v;
        h1b[idx] = (bf16_t)v;
    }
}

// agg2 <- agg2 + h1 @ lw2 + b2   [N,256] in place; each wave does 2 N-tiles
__global__ void __launch_bounds__(256) k_l2(const bf16_t* __restrict__ h1b,
                                            const bf16_t* __restrict__ lw2T,
                                            const float* __restrict__ b2,
                                            float* __restrict__ agg2) {
    const int w     = threadIdx.x >> 5;
    const int lane  = threadIdx.x & 31;
    const int mbase = blockIdx.x * 16;
#pragma unroll
    for (int t = 0; t < 2; ++t) {
        const int nbase = (w * 2 + t) * 16;
        v8f acc = {};
#pragma unroll
        for (int kc = 0; kc < HD; kc += 32) {
            v16bf a = load_a_frag(h1b, HD, mbase, kc);
            v16bf b = load_b_frag(lw2T, HD, nbase, kc);
            acc = __builtin_amdgcn_wmma_f32_16x16x32_bf16(false, a, false, b,
                                                          (short)0, acc, false, false);
        }
        const int col   = nbase + (lane & 15);
        const int rbase = mbase + ((lane >> 4) << 3);
        float bias = b2[col];
#pragma unroll
        for (int r = 0; r < 8; ++r) {
            long idx = (long)(rbase + r) * HD2 + col;
            agg2[idx] += acc[r] + bias;
        }
    }
}

// z = m + sqrt(softplus(hv)+1e-8) * eps   (fp32 for hr, bf16 for decoder B)
__global__ void k_gauss(const float* __restrict__ x2, const float* __restrict__ eps,
                        float* __restrict__ z, bf16_t* __restrict__ zb) {
    int i = blockIdx.x * blockDim.x + threadIdx.x;
    if (i >= N_NODES * HD) return;
    int n = i >> 7, j = i & (HD - 1);
    float m  = x2[(long)n * HD2 + j];
    float hv = x2[(long)n * HD2 + HD + j];
    float sp = (hv > 20.0f) ? hv : log1pf(expf(hv));
    float v  = m + sqrtf(sp + 1e-8f) * eps[i];
    z[i]  = v;
    zb[i] = (bf16_t)v;
}

// hr = z[head] * w_rel[rel]  (bf16, decoder A operand)
__global__ void k_hr(const float* __restrict__ z, const int* __restrict__ head,
                     const int* __restrict__ rel, const float* __restrict__ w_rel,
                     bf16_t* __restrict__ hrb) {
    int i = blockIdx.x * blockDim.x + threadIdx.x;
    if (i >= B_Q * HD) return;
    int b = i >> 7, j = i & (HD - 1);
    hrb[i] = (bf16_t)(z[(long)head[b] * HD + j] * w_rel[(long)rel[b] * HD + j]);
}

// scores = hr @ z^T  -> out [2048, 20000]; zb padded to N_PAD rows so loads are
// unconditional; only the epilogue store is guarded.
__global__ void __launch_bounds__(256) k_dec(const bf16_t* __restrict__ hrb,
                                             const bf16_t* __restrict__ zb,
                                             float* __restrict__ out) {
    const int w     = threadIdx.x >> 5;
    const int lane  = threadIdx.x & 31;
    const int mbase = blockIdx.y * 16;
    const int nbase = blockIdx.x * 128 + w * 16;
    v8f acc = {};
#pragma unroll
    for (int kc = 0; kc < HD; kc += 32) {
        v16bf a = load_a_frag(hrb, HD, mbase, kc);
        v16bf b = load_b_frag(zb, HD, nbase, kc);
        acc = __builtin_amdgcn_wmma_f32_16x16x32_bf16(false, a, false, b,
                                                      (short)0, acc, false, false);
    }
    const int col = nbase + (lane & 15);
    if (col < N_NODES) {
        const int rbase = mbase + ((lane >> 4) << 3);
#pragma unroll
        for (int r = 0; r < 8; ++r)
            out[(long)(rbase + r) * N_NODES + col] = acc[r];
    }
}

// ---------------------------------------------------------------------------
extern "C" void kernel_launch(void* const* d_in, const int* in_sizes, int n_in,
                              void* d_out, int out_size, void* d_ws, size_t ws_size,
                              hipStream_t stream) {
    (void)in_sizes; (void)n_in; (void)out_size; (void)ws_size;
    const int*   h     = (const int*)  d_in[0];
    const int*   src   = (const int*)  d_in[1];
    const int*   dst   = (const int*)  d_in[2];
    const int*   et    = (const int*)  d_in[3];
    const float* norm  = (const float*)d_in[4];
    const int*   headi = (const int*)  d_in[5];
    const int*   reli  = (const int*)  d_in[6];
    const float* embed = (const float*)d_in[7];
    const float* W1    = (const float*)d_in[8];
    const float* lw1   = (const float*)d_in[9];
    const float* b1    = (const float*)d_in[10];
    const float* W2    = (const float*)d_in[11];
    const float* lw2   = (const float*)d_in[12];
    const float* b2    = (const float*)d_in[13];
    const float* w_rel = (const float*)d_in[14];
    const float* eps   = (const float*)d_in[15];
    float* out = (float*)d_out;

    // Workspace layout. fp32 region first, then bf16 region (float-aligned).
    float* ws   = (float*)d_ws;
    float* agg1 = ws;                                   // N*HD    f32
    float* agg2 = agg1 + (long)N_NODES * HD;            // N*HD2   f32
    float* x0   = agg2 + (long)N_NODES * HD2;           // N*HD    f32
    float* h1   = x0   + (long)N_NODES * HD;            // N*HD    f32
    float* z    = h1   + (long)N_NODES * HD;            // N*HD    f32
    bf16_t* bfw  = (bf16_t*)(z + (long)N_NODES * HD);
    bf16_t* x0b  = bfw;                                 // N*HD    bf16
    bf16_t* h1b  = x0b  + (long)N_NODES * HD;           // N*HD    bf16
    bf16_t* zb   = h1b  + (long)N_NODES * HD;           // NPAD*HD bf16 (zero tail)
    bf16_t* hrb  = zb   + (long)N_PAD   * HD;           // B*HD    bf16
    bf16_t* lw1T = hrb  + (long)B_Q * HD;               // HD*HD   bf16 (transposed)
    bf16_t* lw2T = lw1T + HD * HD;                      // HD2*HD  bf16 (transposed)

    k_zero  <<<2048, 256, 0, stream>>>(agg1, (long)N_NODES * (HD + HD2));
    // zero the decoder-B padding rows (96 * 128 bf16 = 6144 floats)
    k_zero  <<<24, 256, 0, stream>>>((float*)(zb + (long)N_NODES * HD),
                                     (long)(N_PAD - N_NODES) * HD / 2);
    k_wT    <<<(HD * HD  + 255) / 256, 256, 0, stream>>>(lw1, lw1T, HD, HD);
    k_wT    <<<(HD * HD2 + 255) / 256, 256, 0, stream>>>(lw2, lw2T, HD, HD2);
    k_gather<<<(N_NODES * HD + 255) / 256, 256, 0, stream>>>(embed, h, x0, x0b);
    k_edge1 <<<N_EDGES / 2, 256, 0, stream>>>(x0, src, dst, et, norm, W1, agg1);
    k_l1    <<<N_NODES / 16, 256, 0, stream>>>(x0b, lw1T, b1, agg1, h1, h1b);
    k_edge2 <<<N_EDGES, 256, 0, stream>>>(h1, src, dst, et, norm, W2, agg2);
    k_l2    <<<N_NODES / 16, 256, 0, stream>>>(h1b, lw2T, b2, agg2);
    k_gauss <<<(N_NODES * HD + 255) / 256, 256, 0, stream>>>(agg2, eps, z, zb);
    k_hr    <<<(B_Q * HD + 255) / 256, 256, 0, stream>>>(z, headi, reli, w_rel, hrb);
    dim3 gdec(N_PAD / 128, B_Q / 16);
    k_dec   <<<gdec, 256, 0, stream>>>(hrb, zb, out);
}
